// GroupQueryAttention_42142219108592
// MI455X (gfx1250) — matrix operand order repaired and checked by reading, not compiled, not run on hardware
//
#include <hip/hip_runtime.h>
#include <hip/hip_bf16.h>
#include <math.h>

// ---------------------------------------------------------------------------
// GQA attention block for MI455X (gfx1250): all matmuls via v_wmma bf16,
// TDM (tensor_load_to_lds) staging for conversion-free tiles.
// B=2, T=2048, D=4096, Hq=32, Hkv=8, Dh=128, G=4.
// ---------------------------------------------------------------------------

typedef __attribute__((ext_vector_type(16))) __bf16    v16bf;
typedef __attribute__((ext_vector_type(8)))  float     v8f;
typedef __attribute__((ext_vector_type(4)))  unsigned  v4u;
typedef __attribute__((ext_vector_type(8)))  int       v8i;
typedef __attribute__((ext_vector_type(4)))  int       v4i;

#if defined(__gfx1250__) && __has_builtin(__builtin_amdgcn_tensor_load_to_lds)
#define HAVE_TDM 1
#else
#define HAVE_TDM 0
#endif

union Frag {
  uint4 q[2];   // 8 VGPRs = 16 bf16
  v16bf v;
};

__device__ __forceinline__ unsigned short f32_to_bf16(float f) {
  unsigned u = __float_as_uint(f);
  u += 0x7FFFu + ((u >> 16) & 1u);          // round-to-nearest-even
  return (unsigned short)(u >> 16);
}
__device__ __forceinline__ float bf16_to_f32(unsigned short h) {
  return __uint_as_float(((unsigned)h) << 16);
}

#if HAVE_TDM
// Issue a 2D TDM load: tile [tile_d1 rows x tile_d0 elems] of 2-byte elements
// from a row-major tensor (row stride = stride0 elems) into LDS at lds_addr,
// optionally padding the LDS destination every row (D# pad feature).
// Call from ONE wave only (EXEC ignored; whole-wave DMA). Track via TENSORcnt.
// This toolchain (clang-23 lane) uses the 6-arg builtin:
//   (uint32x4 g0, int32x8 g1, int32x4 g2, int32x4 g3, int32x8 gx, i32 cpol)
__device__ __forceinline__ void tdm_load_2d_bf16(
    unsigned lds_addr, const void* gptr,
    unsigned tile_d0, unsigned tile_d1,
    unsigned tensor_d0, unsigned tensor_d1, unsigned stride0,
    unsigned pad_interval_code, unsigned pad_amount_code, int pad_en) {
  unsigned long long ga = (unsigned long long)(size_t)gptr;
  v4u g0;
  g0[0] = 1u;                                           // count=1, user desc
  g0[1] = lds_addr;                                     // [63:32]  lds_addr
  g0[2] = (unsigned)ga;                                 // [95:64]  global lo
  g0[3] = (unsigned)(ga >> 32) | (2u << 30);            // [120:96] hi, type=2
  v8i g1;
  unsigned e0 = (1u << 16);                             // data_size = 2 bytes
  if (pad_en) e0 |= (1u << 20) | (pad_interval_code << 22) | (pad_amount_code << 25);
  g1[0] = (int)e0;                                      // wg_mask=0, flags
  g1[1] = (int)(tensor_d0 << 16);                       // tensor_dim0 [79:48]
  g1[2] = (int)((tensor_d0 >> 16) | (tensor_d1 << 16)); // + tensor_dim1 [111:80]
  g1[3] = (int)((tensor_d1 >> 16) | (tile_d0 << 16));   // + tile_dim0 [127:112]
  g1[4] = (int)tile_d1;                                 // tile_dim1, tile_dim2=0
  g1[5] = (int)stride0;                                 // tensor_dim0_stride lo
  g1[6] = 0;                                            // stride0 hi, stride1 lo
  g1[7] = 0;
  v4i gz4 = {0, 0, 0, 0};                               // 2D: groups 2/3 unused
  v8i gz8 = {0, 0, 0, 0, 0, 0, 0, 0};
  __builtin_amdgcn_tensor_load_to_lds(g0, g1, gz4, gz4, gz8, 0);
}
#endif

// A-matrix fragment, 16x32 bf16 (M x K), source row-major [row][k], ld elems.
// ISA layout: lanes 0-15 -> M, half0 holds K 0..7 & 16..23, half1 K 8..15 & 24..31.
__device__ __forceinline__ void load_frag_a(Frag& f, const unsigned short* base,
                                            int row0, int ld, int k0, int lane) {
  int m = lane & 15, h = lane >> 4;
  const unsigned short* p = base + (size_t)(row0 + m) * ld + k0 + h * 8;
  f.q[0] = *(const uint4*)(p);
  f.q[1] = *(const uint4*)(p + 16);
}
// B-matrix fragment, 32x16 bf16 (K x N), source stored K-contiguous per column:
// base[col][k], ld elems. Lanes 0-15 hold K 0..7 & 16..23, lanes 16-31 K 8..15 & 24..31.
__device__ __forceinline__ void load_frag_b(Frag& f, const unsigned short* base,
                                            int col0, int ld, int k0, int lane) {
  int n = lane & 15, h = lane >> 4;
  const unsigned short* p = base + (size_t)(col0 + n) * ld + k0 + h * 8;
  f.q[0] = *(const uint4*)(p);
  f.q[1] = *(const uint4*)(p + 16);
}

// ---------------------------------------------------------------------------
// GEMM: C[M,N] = A[M,K] @ B[K,N] (+bias). A fp32 or bf16 (bf16 path staged by
// TDM), B fp32 converted in staging. mode 0: fp32 row-major out.
// mode 1: bf16 head-major [B,H,2048,128]. 256 threads, tile 128x128, K-step 64.
// ---------------------------------------------------------------------------
__global__ __launch_bounds__(256) void gemm_wmma_kernel(
    const void* __restrict__ Ap, int a_is_bf16,
    const float* __restrict__ Bp, const float* __restrict__ bias,
    unsigned short* __restrict__ outH, float* __restrict__ outF,
    int M, int N, int K, int mode, int H) {
  __shared__ __align__(16) unsigned short As[128][72];  // [m][k], pad 64->72
  __shared__ __align__(16) unsigned short Bs[128][72];  // [n][k] (transposed)

  const int tid = threadIdx.x;
  const int lane = tid & 31, wave = tid >> 5;
  const int wm = wave >> 1, wn = wave & 1;              // 4x2 wave grid
  const int m0 = blockIdx.y * 128, n0 = blockIdx.x * 128;

  const v8f vzero = {0.f,0.f,0.f,0.f,0.f,0.f,0.f,0.f};
  v8f acc[2][4];
  for (int i = 0; i < 2; ++i)
    for (int j = 0; j < 4; ++j) acc[i][j] = vzero;

  for (int kk = 0; kk < K; kk += 64) {
    // ---- stage A tile [128 x 64] into LDS
    if (a_is_bf16) {
#if HAVE_TDM
      if (wave == 0) {
        const unsigned short* A = (const unsigned short*)Ap;
        // rows of 64 bf16 = 32 DWORDs (interval code 4), pad 4 DWORDs (code 3)
        tdm_load_2d_bf16((unsigned)(size_t)&As[0][0],
                         &A[(size_t)m0 * K + kk],
                         /*tile_d0=*/64, /*tile_d1=*/128,
                         /*tensor_d0=*/(unsigned)K, /*tensor_d1=*/(unsigned)M,
                         /*stride0=*/(unsigned)K, 4, 3, 1);
      }
#else
      const unsigned short* A = (const unsigned short*)Ap;
      for (int i = tid; i < 128 * 8; i += 256) {
        int row = i >> 3, c = (i & 7) * 8;
        *(uint4*)&As[row][c] = *(const uint4*)&A[(size_t)(m0 + row) * K + kk + c];
      }
#endif
    } else {
      const float* A = (const float*)Ap;
      for (int i = tid; i < 128 * 8; i += 256) {
        int row = i >> 3, c = (i & 7) * 8;
        const float* s = &A[(size_t)(m0 + row) * K + kk + c];
        uint4 d;
        d.x = (unsigned)f32_to_bf16(s[0]) | ((unsigned)f32_to_bf16(s[1]) << 16);
        d.y = (unsigned)f32_to_bf16(s[2]) | ((unsigned)f32_to_bf16(s[3]) << 16);
        d.z = (unsigned)f32_to_bf16(s[4]) | ((unsigned)f32_to_bf16(s[5]) << 16);
        d.w = (unsigned)f32_to_bf16(s[6]) | ((unsigned)f32_to_bf16(s[7]) << 16);
        *(uint4*)&As[row][c] = d;
      }
    }
    // ---- stage B tile [64 x 128] transposed -> Bs[n][k], fp32->bf16
    for (int i = tid; i < 64 * 16; i += 256) {
      int r = i >> 4, cc = (i & 15) * 8;
      const float* s = &Bp[(size_t)(kk + r) * N + n0 + cc];
#pragma unroll
      for (int e = 0; e < 8; ++e) Bs[cc + e][r] = f32_to_bf16(s[e]);
    }
    if (kk + 64 < K)   // global_prefetch_b8 of next B slab
      __builtin_prefetch(&Bp[(size_t)(kk + 64 + (tid >> 2)) * N + n0 + (tid & 3) * 32], 0, 1);
#if HAVE_TDM
    if (a_is_bf16 && wave == 0) __builtin_amdgcn_s_wait_tensorcnt(0);
#endif
    __syncthreads();

#pragma unroll
    for (int ks = 0; ks < 64; ks += 32) {
      Frag a0, a1;
      load_frag_a(a0, &As[0][0], wm * 32,      72, ks, lane);
      load_frag_a(a1, &As[0][0], wm * 32 + 16, 72, ks, lane);
#pragma unroll
      for (int j = 0; j < 4; ++j) {
        Frag bfr;
        load_frag_b(bfr, &Bs[0][0], wn * 64 + j * 16, 72, ks, lane);
        acc[0][j] = __builtin_amdgcn_wmma_f32_16x16x32_bf16(
            false, a0.v, false, bfr.v, (short)0, acc[0][j], false, false);
        acc[1][j] = __builtin_amdgcn_wmma_f32_16x16x32_bf16(
            false, a1.v, false, bfr.v, (short)0, acc[1][j], false, false);
      }
    }
    __syncthreads();
  }

  // ---- epilogue. C layout: vgpr r, lane l -> (m = r + 8*(l>>4), n = l&15)
  const int hl = lane >> 4, nlo = lane & 15;
  for (int i = 0; i < 2; ++i)
    for (int j = 0; j < 4; ++j)
#pragma unroll
      for (int r = 0; r < 8; ++r) {
        int mrow = m0 + wm * 32 + i * 16 + r + 8 * hl;
        int ncol = n0 + wn * 64 + j * 16 + nlo;
        float v = acc[i][j][r];
        if (bias) v += bias[ncol];
        if (mode == 0) {
          outF[(size_t)mrow * N + ncol] = v;
        } else {
          int bb = mrow >> 11, t = mrow & 2047;    // m = b*2048 + t
          int hh = ncol >> 7,  d = ncol & 127;     // n = h*128 + d
          outH[(((size_t)bb * H + hh) * 2048 + t) * 128 + d] = f32_to_bf16(v);
        }
      }
}

// ---------------------------------------------------------------------------
// RoPE (rotate-half) in place on bf16 [B,H,2048,128]; optionally emit fp32 copy
// (post-rope present_k). One thread per (b,h,t,d<64).
// ---------------------------------------------------------------------------
__global__ void rope_kernel(unsigned short* __restrict__ buf,
                            float* __restrict__ fout, int total) {
  int idx = blockIdx.x * blockDim.x + threadIdx.x;
  if (idx >= total) return;
  int d  = idx & 63;
  int t  = (idx >> 6) & 2047;
  int bh = idx >> 17;
  float freq = __expf(-(float)d * (9.210340371976184f / 64.0f)); // 10000^(-2d/128)
  float ang = (float)t * freq;
  float sv, cv;
  sincosf(ang, &sv, &cv);
  size_t base = ((size_t)bh * 2048 + t) * 128;
  float x1 = bf16_to_f32(buf[base + d]);
  float x2 = bf16_to_f32(buf[base + d + 64]);
  float r1 = x1 * cv - x2 * sv;
  float r2 = x2 * cv + x1 * sv;
  buf[base + d]      = f32_to_bf16(r1);
  buf[base + d + 64] = f32_to_bf16(r2);
  if (fout) { fout[base + d] = r1; fout[base + d + 64] = r2; }
}

__global__ void bf16_to_f32_kernel(const unsigned short* __restrict__ src,
                                   float* __restrict__ dst, int n) {
  int i = blockIdx.x * blockDim.x + threadIdx.x;
  if (i < n) dst[i] = bf16_to_f32(src[i]);
}

// ---------------------------------------------------------------------------
// Flash attention (causal, GQA). 256 threads = 8 waves; one block per
// (b, hq, 128-row q tile); each wave owns 16 query rows; 64-key iterations.
// K tile staged by TDM (with LDS pad), V staged transposed manually.
// ---------------------------------------------------------------------------
__global__ __launch_bounds__(256) void attn_kernel(
    const unsigned short* __restrict__ qb, const unsigned short* __restrict__ kb,
    const unsigned short* __restrict__ vb, unsigned short* __restrict__ ob) {
  __shared__ __align__(16) unsigned short Ks[64][136];   // [key][feat]
  __shared__ __align__(16) unsigned short Vt[128][72];   // [feat][key]
  __shared__ __align__(16) unsigned short Ps[8][16][72]; // per-wave P tile

  const int tid = threadIdx.x, lane = tid & 31, wave = tid >> 5;
  const int qt = blockIdx.x, hq = blockIdx.y, b = blockIdx.z;
  const int hkv = hq >> 2;
  const int q0 = qt * 128;
  const size_t qbase  = (((size_t)b * 32 + hq) * 2048 + q0) * 128;
  const size_t kvbase = ((size_t)b * 8 + hkv) * 2048 * 128;
  const int hl = lane >> 4, nlo = lane & 15;

  // Q fragments for this wave's 16 rows, whole Dh=128, kept in registers.
  Frag aq[4];
#pragma unroll
  for (int kc = 0; kc < 4; ++kc)
    load_frag_a(aq[kc], qb + qbase, wave * 16, 128, kc * 32, lane);

  const v8f vzero = {0.f,0.f,0.f,0.f,0.f,0.f,0.f,0.f};
  v8f o[8];
  float m_i[8], l_i[8];
#pragma unroll
  for (int f = 0; f < 8; ++f) o[f] = vzero;
#pragma unroll
  for (int r = 0; r < 8; ++r) { m_i[r] = -3.0e38f; l_i[r] = 0.f; }

  const float scale = 0.08838834764831845f; // 1/sqrt(128)
  const int nk = (q0 + 128) / 64;           // causal: keys < q0+128

  for (int it = 0; it < nk; ++it) {
    const int key0 = it * 64;
    // ---- stage K [64 x 128] -> Ks (padded rows)
#if HAVE_TDM
    if (wave == 0) {
      // rows of 128 bf16 = 64 DWORDs (interval code 5), pad 4 DWORDs (code 3)
      tdm_load_2d_bf16((unsigned)(size_t)&Ks[0][0],
                       kb + kvbase + (size_t)key0 * 128,
                       /*tile_d0=*/128, /*tile_d1=*/64,
                       /*tensor_d0=*/128, /*tensor_d1=*/2048,
                       /*stride0=*/128, 5, 3, 1);
    }
#else
    for (int i = tid; i < 64 * 16; i += 256) {
      int row = i >> 4, c = (i & 15) * 8;
      *(uint4*)&Ks[row][c] =
          *(const uint4*)&kb[kvbase + (size_t)(key0 + row) * 128 + c];
    }
#endif
    // ---- stage V transposed -> Vt[feat][key]
    for (int i = tid; i < 64 * 16; i += 256) {
      int row = i >> 4, c = (i & 15) * 8;
      uint4 u = *(const uint4*)&vb[kvbase + (size_t)(key0 + row) * 128 + c];
      unsigned short tmp[8];
      *(uint4*)tmp = u;
#pragma unroll
      for (int e = 0; e < 8; ++e) Vt[c + e][row] = tmp[e];
    }
#if HAVE_TDM
    if (wave == 0) __builtin_amdgcn_s_wait_tensorcnt(0);
#endif
    __syncthreads();

    // ---- S = Q K^T : [16 q x 64 keys] in 4 accumulators
    v8f s4[4];
#pragma unroll
    for (int j = 0; j < 4; ++j) s4[j] = vzero;
#pragma unroll
    for (int j = 0; j < 4; ++j)
#pragma unroll
      for (int kc = 0; kc < 4; ++kc) {
        Frag bk;
        load_frag_b(bk, &Ks[0][0], j * 16, 136, kc * 32, lane);
        s4[j] = __builtin_amdgcn_wmma_f32_16x16x32_bf16(
            false, aq[kc].v, false, bk.v, (short)0, s4[j], false, false);
      }

    // ---- scale + causal mask
#pragma unroll
    for (int j = 0; j < 4; ++j)
#pragma unroll
      for (int r = 0; r < 8; ++r) {
        int qrow = q0 + wave * 16 + r + 8 * hl;
        int key  = key0 + j * 16 + nlo;
        float sv = s4[j][r] * scale;
        if (key > qrow) sv = -3.0e38f;
        s4[j][r] = sv;
      }

    // ---- online softmax (row lives in one 16-lane half; xor 1/2/4/8 stays in it)
    float alpha[8], mnew[8];
#pragma unroll
    for (int r = 0; r < 8; ++r) {
      float rm = fmaxf(fmaxf(s4[0][r], s4[1][r]), fmaxf(s4[2][r], s4[3][r]));
      rm = fmaxf(rm, __shfl_xor(rm, 1, 32));
      rm = fmaxf(rm, __shfl_xor(rm, 2, 32));
      rm = fmaxf(rm, __shfl_xor(rm, 4, 32));
      rm = fmaxf(rm, __shfl_xor(rm, 8, 32));
      mnew[r]  = fmaxf(m_i[r], rm);
      alpha[r] = __expf(m_i[r] - mnew[r]);
      m_i[r]   = mnew[r];
    }
#pragma unroll
    for (int r = 0; r < 8; ++r) {
      float ps = 0.f;
#pragma unroll
      for (int j = 0; j < 4; ++j) {
        float p = __expf(s4[j][r] - mnew[r]);
        Ps[wave][r + 8 * hl][j * 16 + nlo] = f32_to_bf16(p);
        ps += p;
      }
      ps += __shfl_xor(ps, 1, 32);
      ps += __shfl_xor(ps, 2, 32);
      ps += __shfl_xor(ps, 4, 32);
      ps += __shfl_xor(ps, 8, 32);
      l_i[r] = l_i[r] * alpha[r] + ps;
    }
#pragma unroll
    for (int f = 0; f < 8; ++f)
#pragma unroll
      for (int r = 0; r < 8; ++r) o[f][r] *= alpha[r];
    __syncthreads();   // publish Ps (cross-lane) before fragment reads

    // ---- O += P V
    Frag ap[2];
    load_frag_a(ap[0], &Ps[wave][0][0], 0, 72, 0, lane);
    load_frag_a(ap[1], &Ps[wave][0][0], 0, 72, 32, lane);
#pragma unroll
    for (int f = 0; f < 8; ++f)
#pragma unroll
      for (int kj = 0; kj < 2; ++kj) {
        Frag bv;
        load_frag_b(bv, &Vt[0][0], f * 16, 72, kj * 32, lane);
        o[f] = __builtin_amdgcn_wmma_f32_16x16x32_bf16(
            false, ap[kj].v, false, bv.v, (short)0, o[f], false, false);
      }
    __syncthreads();   // Ks/Vt reused next iteration
  }

  // ---- normalize & store: ob is [B, T, 4096] bf16, column block hq*128
  float inv[8];
#pragma unroll
  for (int r = 0; r < 8; ++r) inv[r] = 1.0f / l_i[r];
  const size_t obase = ((size_t)b * 2048 + q0) * 4096 + (size_t)hq * 128;
#pragma unroll
  for (int f = 0; f < 8; ++f)
#pragma unroll
    for (int r = 0; r < 8; ++r) {
      int trow = wave * 16 + r + 8 * hl;
      int d    = f * 16 + nlo;
      ob[obase + (size_t)trow * 4096 + d] = f32_to_bf16(o[f][r] * inv[r]);
    }
}

// ---------------------------------------------------------------------------
extern "C" void kernel_launch(void* const* d_in, const int* in_sizes, int n_in,
                              void* d_out, int out_size, void* d_ws, size_t ws_size,
                              hipStream_t stream) {
  const float* x  = (const float*)d_in[0];
  const float* Wq = (const float*)d_in[1];
  const float* bq = (const float*)d_in[2];
  const float* Wk = (const float*)d_in[3];
  const float* bk = (const float*)d_in[4];
  const float* Wv = (const float*)d_in[5];
  const float* bv = (const float*)d_in[6];
  const float* Wo = (const float*)d_in[7];
  float* out = (float*)d_out;

  // workspace: bf16 q[2,32,2048,128], k/v[2,8,2048,128], attn-out[2,2048,4096]
  const size_t QB = 16777216, KVB = 4194304;
  unsigned short* qb = (unsigned short*)d_ws;
  unsigned short* kb = qb + QB;
  unsigned short* vb = kb + KVB;
  unsigned short* ob = vb + KVB;

  float* out_y = out;                         // [2,2048,4096]
  float* out_k = out + 16777216;              // [2,8,2048,128] (post-rope)
  float* out_v = out + 16777216 + 4194304;    // [2,8,2048,128]

  dim3 blk(256);
  // QKV projections (fp32 in, bf16 head-major out)
  gemm_wmma_kernel<<<dim3(32, 32), blk, 0, stream>>>(x, 0, Wq, bq, qb, nullptr,
                                                     4096, 4096, 4096, 1, 32);
  gemm_wmma_kernel<<<dim3(8, 32), blk, 0, stream>>>(x, 0, Wk, bk, kb, nullptr,
                                                    4096, 1024, 4096, 1, 8);
  gemm_wmma_kernel<<<dim3(8, 32), blk, 0, stream>>>(x, 0, Wv, bv, vb, nullptr,
                                                    4096, 1024, 4096, 1, 8);
  // RoPE (q in place; k in place + fp32 present_k), present_v copy
  rope_kernel<<<dim3(32768), blk, 0, stream>>>(qb, nullptr, 2 * 32 * 2048 * 64);
  rope_kernel<<<dim3(8192), blk, 0, stream>>>(kb, out_k, 2 * 8 * 2048 * 64);
  bf16_to_f32_kernel<<<dim3(16384), blk, 0, stream>>>(vb, out_v, 4194304);
  // causal GQA flash attention
  attn_kernel<<<dim3(16, 32, 2), blk, 0, stream>>>(qb, kb, vb, ob);
  // output projection (bf16 A staged via TDM, fp32 out)
  gemm_wmma_kernel<<<dim3(32, 32), blk, 0, stream>>>(ob, 1, Wo, nullptr, nullptr,
                                                     out_y, 4096, 4096, 4096, 0, 32);
  (void)in_sizes; (void)n_in; (void)out_size; (void)ws_size;
}
